// Msingi1_12412455485565
// MI455X (gfx1250) — matrix operand order repaired and checked by reading, not compiled, not run on hardware
//
#include <hip/hip_runtime.h>
#include <cstddef>

// ---------------- model dims ----------------
#define VOCAB 32000
#define HDIM  1024
#define NHEAD 16
#define DHEAD 64
#define IDIM  4096
#define NLAYER 8
#define BATCH 2
#define SEQ   2048
#define NTOK  (BATCH*SEQ)   // 4096
#define EPS   1e-5f

typedef __attribute__((ext_vector_type(16))) __bf16 v16bf;
typedef __attribute__((ext_vector_type(8)))  __bf16 bf16x8;
typedef __attribute__((ext_vector_type(8)))  float  v8f;

// ---------------- helpers ----------------
__device__ inline v8f vzero8() {
    v8f z;
#pragma unroll
    for (int i = 0; i < 8; ++i) z[i] = 0.0f;
    return z;
}

__device__ inline v8f wmma_bf16(v16bf a, v16bf b, v8f c) {
    // D = A(16x32 bf16) x B(32x16 bf16) + C(16x16 f32)
    return __builtin_amdgcn_wmma_f32_16x16x32_bf16(
        false, a, false, b, (short)0, c, false, false);
}

// two groups of 8 consecutive bf16 (16B aligned, LDS or global) -> v16bf frag
__device__ inline v16bf frag_pair(const __bf16* p0, const __bf16* p1) {
    bf16x8 lo = *(const bf16x8*)p0;
    bf16x8 hi = *(const bf16x8*)p1;
    return __builtin_shufflevector(lo, hi, 0,1,2,3,4,5,6,7,8,9,10,11,12,13,14,15);
}

__device__ inline bf16x8 cvt8(float4 a, float4 b) {
    bf16x8 o;
    o[0]=(__bf16)a.x; o[1]=(__bf16)a.y; o[2]=(__bf16)a.z; o[3]=(__bf16)a.w;
    o[4]=(__bf16)b.x; o[5]=(__bf16)b.y; o[6]=(__bf16)b.z; o[7]=(__bf16)b.w;
    return o;
}

// async copy 16B global -> LDS (ASYNCcnt-tracked), GVS addressing
__device__ inline void async_cp16(unsigned lds_off, unsigned glb_off, const void* base) {
    asm volatile("global_load_async_to_lds_b128 %0, %1, %2"
                 :: "v"(lds_off), "v"(glb_off),
                    "s"((unsigned long long)base)
                 : "memory");
}
__device__ inline void wait_async0() {
    asm volatile("s_wait_asynccnt 0x0" ::: "memory");
}

// ---------------- fp32 -> bf16 bulk convert (weights, once per launch) ----------------
__global__ __launch_bounds__(256) void cvt_kernel(
    const float* __restrict__ in, __bf16* __restrict__ out)
{
    size_t i = ((size_t)blockIdx.x * 256 + threadIdx.x) * 8;
    float4 a = *(const float4*)(in + i);
    float4 b = *(const float4*)(in + i + 4);
    *(bf16x8*)(out + i) = cvt8(a, b);
}

// ---------------- embedding (writes fp32 h + bf16 shadow) ----------------
__global__ __launch_bounds__(256) void embed_kernel(
    const int* __restrict__ ids, const float* __restrict__ tok,
    const float* __restrict__ pos, float* __restrict__ h, __bf16* __restrict__ hb)
{
    int t = blockIdx.x;
    int s = t & (SEQ - 1);
    int id = ids[t];
    const float* tr = tok + (size_t)id * HDIM;
    const float* pr = pos + (size_t)s * HDIM;
    float*  hr = h  + (size_t)t * HDIM;
    __bf16* br = hb + (size_t)t * HDIM;
    for (int i = threadIdx.x; i < HDIM; i += 256) {
        float v = tr[i] + pr[i];
        hr[i] = v;
        br[i] = (__bf16)v;
    }
}

// ---------------- GEMM: C[M,N] = A[M,K](bf16) @ W[N,K](bf16)^T + bias ----------------
// 64x128 block tile, K-tile 64 (2 WMMA k-steps), 256 thr = 8 waves x (2x2 tiles).
// Double-buffered LDS staged with GLOBAL_LOAD_ASYNC_TO_LDS_B128:
// 8 WMMAs per wait+barrier round, copies for tile k+1 in flight during compute.
#define LDSA 72   // 64 + 8 pad (bf16), 144B row stride (conflict-free b128 reads)
#define LDSB 72

template<int ACT, typename OT>   // ACT: 0 none, 1 exact GELU; OT: float or __bf16
__global__ __launch_bounds__(256) void gemm_kernel(
    const __bf16* __restrict__ A, const __bf16* __restrict__ W,
    const float* __restrict__ bias, OT* __restrict__ C,
    int M, int N, int K)
{
    __shared__ __align__(16) __bf16 As[2 * 64 * LDSA];
    __shared__ __align__(16) __bf16 Bs[2 * 128 * LDSB];

    const int tid  = threadIdx.x;
    const int lane = tid & 31;
    const int wv   = tid >> 5;
    const int l15  = lane & 15;
    const int half = lane >> 4;

    const int bm = blockIdx.y * 64;
    const int bn = blockIdx.x * 128;
    const int wm = (wv >> 2) * 32;
    const int wn = (wv & 3) * 32;

    const unsigned asb = (unsigned)(size_t)&As[0];
    const unsigned bsb = (unsigned)(size_t)&Bs[0];

    v8f acc[2][2];
    acc[0][0] = vzero8(); acc[0][1] = vzero8();
    acc[1][0] = vzero8(); acc[1][1] = vzero8();

    // stage one 64x64 A tile (8KB: 2 chunks/thr) + 128x64 W tile (16KB: 4 chunks/thr)
    auto stage = [&](int kt, int buf) {
#pragma unroll
        for (int j = 0; j < 2; ++j) {
            int lin = tid + j * 256;
            int row = lin >> 3, c0 = (lin & 7) * 8;
            unsigned lds = asb + (unsigned)((buf * 64 * LDSA + row * LDSA + c0) * 2);
            unsigned gof = (unsigned)(((size_t)(bm + row) * K + kt + c0) * 2);
            async_cp16(lds, gof, A);
        }
#pragma unroll
        for (int j = 0; j < 4; ++j) {
            int lin = tid + j * 256;
            int row = lin >> 3, c0 = (lin & 7) * 8;
            unsigned lds = bsb + (unsigned)((buf * 128 * LDSB + row * LDSB + c0) * 2);
            unsigned gof = (unsigned)(((size_t)(bn + row) * K + kt + c0) * 2);
            async_cp16(lds, gof, W);
        }
    };

    stage(0, 0);
    int buf = 0;
    const int k0a = half ? 8 : 0;


    for (int kt = 0; kt < K; kt += 64) {
        wait_async0();          // this buffer's copies (issued last round) landed
        __syncthreads();        // ...for every wave; prev compute on other buf done
        if (kt + 64 < K) stage(kt + 64, buf ^ 1);   // overlap next copy with WMMA

        const __bf16* ab = &As[buf * 64 * LDSA];
        const __bf16* bb = &Bs[buf * 128 * LDSB];
#pragma unroll
        for (int ks = 0; ks < 2; ++ks) {            // two 32-wide k-steps per tile
            v16bf af[2], bf[2];
#pragma unroll
            for (int mt = 0; mt < 2; ++mt) {
                const __bf16* p = &ab[(wm + mt * 16 + l15) * LDSA + ks * 32 + k0a];
                af[mt] = frag_pair(p, p + 16);
            }
#pragma unroll
            for (int nt = 0; nt < 2; ++nt) {
                const __bf16* p = &bb[(wn + nt * 16 + l15) * LDSB + ks * 32 + k0a];
                bf[nt] = frag_pair(p, p + 16);
            }
#pragma unroll
            for (int mt = 0; mt < 2; ++mt)
#pragma unroll
                for (int nt = 0; nt < 2; ++nt)
                    acc[mt][nt] = wmma_bf16(af[mt], bf[nt], acc[mt][nt]);
        }
        buf ^= 1;
    }

    // epilogue: bias / GELU, store fp32 or bf16
#pragma unroll
    for (int mt = 0; mt < 2; ++mt) {
#pragma unroll
        for (int nt = 0; nt < 2; ++nt) {
            int gn = bn + wn + nt * 16 + l15;
            float bv = bias ? bias[gn] : 0.0f;
#pragma unroll
            for (int r = 0; r < 8; ++r) {
                int gm = bm + wm + mt * 16 + r + half * 8;
                float v = acc[mt][nt][r] + bv;
                if (ACT == 1) v = 0.5f * v * (1.0f + erff(v * 0.70710678118654752f));
                C[(size_t)gm * N + gn] = (OT)v;
            }
        }
    }
}

// ---------------- flash attention (q,k,v bf16; ctx fp32) ----------------
#define LDSK 72   // 64 + 8 pad
#define LDSV 40   // 32 + 8 pad
#define LDSP 40

__global__ __launch_bounds__(128) void attn_kernel(
    const __bf16* __restrict__ q, const __bf16* __restrict__ k,
    const __bf16* __restrict__ v, const float* __restrict__ am,
    float* __restrict__ ctx)
{
    __shared__ __align__(16) __bf16 Ks [32 * LDSK];   // [kp][d]
    __shared__ __align__(16) __bf16 Vst[64 * LDSV];   // [d][kp] transposed
    __shared__ __align__(16) __bf16 Ps [64 * LDSP];   // 4 waves x 16 rows x 32

    const int bh = blockIdx.x;
    const int b  = bh >> 4, hh = bh & 15;
    const int w    = threadIdx.x >> 5;
    const int lane = threadIdx.x & 31;
    const int l15  = lane & 15;
    const int half = lane >> 4;
    const int q0   = blockIdx.y * 64 + w * 16;
    const float scale = 0.125f;   // 1/sqrt(64)
    const unsigned ksb = (unsigned)(size_t)&Ks[0];

    // q fragments, reused over all kp chunks
    const __bf16* qrow = q + ((size_t)(b * SEQ + q0 + l15)) * HDIM + hh * DHEAD;
    const int k0a = half ? 8 : 0;
    v16bf aq[2];
#pragma unroll
    for (int i = 0; i < 2; ++i) {
        const __bf16* p = qrow + i * 32 + k0a;
        aq[i] = frag_pair(p, p + 16);
    }

    float mrow[8], lrow[8];
#pragma unroll
    for (int r = 0; r < 8; ++r) { mrow[r] = -1.0e30f; lrow[r] = 0.0f; }
    v8f c[4];
#pragma unroll
    for (int t = 0; t < 4; ++t) c[t] = vzero8();

    for (int kp0 = 0; kp0 < SEQ; kp0 += 32) {
        __syncthreads();
        // async-stage K chunk [32 kp][64 d] (4KB: 2 x 16B chunks per thread)
#pragma unroll
        for (int j = 0; j < 2; ++j) {
            int lin = threadIdx.x + j * 128;
            int row = lin >> 3, c0 = (lin & 7) * 8;
            unsigned lds = ksb + (unsigned)((row * LDSK + c0) * 2);
            unsigned gof = (unsigned)((((size_t)(b * SEQ + kp0 + row)) * HDIM + hh * DHEAD + c0) * 2);
            async_cp16(lds, gof, k);
        }
        {   // stage V chunk transposed: Vst[d][kp]
            int row = threadIdx.x >> 2, d0 = (threadIdx.x & 3) * 16;
            const __bf16* gv = v + ((size_t)(b * SEQ + kp0 + row)) * HDIM + hh * DHEAD + d0;
            bf16x8 v0 = *(const bf16x8*)gv;
            bf16x8 v1 = *(const bf16x8*)(gv + 8);
#pragma unroll
            for (int j = 0; j < 8; ++j) {
                Vst[(d0 + j)     * LDSV + row] = v0[j];
                Vst[(d0 + 8 + j) * LDSV + row] = v1[j];
            }
        }
        wait_async0();
        __syncthreads();

        // scores S(16x32): two 16x16 n-tiles, DH=64 -> 2 k-steps each
        v8f sacc[2];
#pragma unroll
        for (int nt = 0; nt < 2; ++nt) {
            sacc[nt] = vzero8();
#pragma unroll
            for (int i = 0; i < 2; ++i) {
                const __bf16* p = &Ks[(nt * 16 + l15) * LDSK + i * 32 + k0a];
                v16bf kf = frag_pair(p, p + 16);
                sacc[nt] = wmma_bf16(aq[i], kf, sacc[nt]);
            }
            int kpg = kp0 + nt * 16 + l15;
            float madd = (1.0f - am[b * SEQ + kpg]) * -10000.0f;
#pragma unroll
            for (int r = 0; r < 8; ++r)
                sacc[nt][r] = sacc[nt][r] * scale + madd;
        }

        // online softmax per row (rows spread over vgpr r + lane-half)
#pragma unroll
        for (int r = 0; r < 8; ++r) {
            float t0 = fmaxf(sacc[0][r], sacc[1][r]);
            t0 = fmaxf(t0, __shfl_xor(t0, 1, 16));
            t0 = fmaxf(t0, __shfl_xor(t0, 2, 16));
            t0 = fmaxf(t0, __shfl_xor(t0, 4, 16));
            t0 = fmaxf(t0, __shfl_xor(t0, 8, 16));
            float mn = fmaxf(mrow[r], t0);
            float alpha = expf(mrow[r] - mn);
            float p0 = expf(sacc[0][r] - mn);
            float p1 = expf(sacc[1][r] - mn);
            float rs = p0 + p1;
            rs += __shfl_xor(rs, 1, 16);
            rs += __shfl_xor(rs, 2, 16);
            rs += __shfl_xor(rs, 4, 16);
            rs += __shfl_xor(rs, 8, 16);
            lrow[r] = lrow[r] * alpha + rs;
            mrow[r] = mn;
#pragma unroll
            for (int t = 0; t < 4; ++t) c[t][r] = c[t][r] * alpha;
            int prow = w * 16 + r + half * 8;
            Ps[prow * LDSP + l15]      = (__bf16)p0;
            Ps[prow * LDSP + 16 + l15] = (__bf16)p1;
        }
        asm volatile("s_wait_dscnt 0x0" ::: "memory");   // wave-local P RAW

        // ctx += P(16x32) @ V(32x64)
        const __bf16* pp = &Ps[(w * 16 + l15) * LDSP + k0a];
        v16bf pa = frag_pair(pp, pp + 16);
#pragma unroll
        for (int t = 0; t < 4; ++t) {
            const __bf16* pv = &Vst[(t * 16 + l15) * LDSV + k0a];
            v16bf vf = frag_pair(pv, pv + 16);
            c[t] = wmma_bf16(pa, vf, c[t]);
        }
    }

    // normalize + write ctx [B,S,H] fp32
#pragma unroll
    for (int t = 0; t < 4; ++t) {
#pragma unroll
        for (int r = 0; r < 8; ++r) {
            int row = q0 + r + half * 8;
            int d = t * 16 + l15;
            ctx[((size_t)(b * SEQ + row)) * HDIM + hh * DHEAD + d] = c[t][r] / lrow[r];
        }
    }
}

// ---------------- residual add + LayerNorm (fp32 out + bf16 shadow) ----------------
__global__ __launch_bounds__(256) void addln_kernel(
    const float* __restrict__ x, const float* __restrict__ y,
    const float* __restrict__ g, const float* __restrict__ bta,
    float* __restrict__ out, __bf16* __restrict__ outb)
{
    __shared__ float sred[256];
    int t = blockIdx.x;
    const float* xr = x + (size_t)t * HDIM;
    const float* yr = y ? y + (size_t)t * HDIM : nullptr;
    float val[4];
    float s = 0.0f;
#pragma unroll
    for (int j = 0; j < 4; ++j) {
        int i = threadIdx.x + j * 256;
        val[j] = xr[i] + (yr ? yr[i] : 0.0f);
        s += val[j];
    }
    sred[threadIdx.x] = s; __syncthreads();
    for (int o = 128; o > 0; o >>= 1) {
        if (threadIdx.x < o) sred[threadIdx.x] += sred[threadIdx.x + o];
        __syncthreads();
    }
    float mean = sred[0] * (1.0f / HDIM);
    __syncthreads();
    float vs = 0.0f;
#pragma unroll
    for (int j = 0; j < 4; ++j) { float d = val[j] - mean; vs += d * d; }
    sred[threadIdx.x] = vs; __syncthreads();
    for (int o = 128; o > 0; o >>= 1) {
        if (threadIdx.x < o) sred[threadIdx.x] += sred[threadIdx.x + o];
        __syncthreads();
    }
    float rstd = rsqrtf(sred[0] * (1.0f / HDIM) + EPS);
    float*  orow = out  + (size_t)t * HDIM;
    __bf16* brow = outb + (size_t)t * HDIM;
#pragma unroll
    for (int j = 0; j < 4; ++j) {
        int i = threadIdx.x + j * 256;
        float o = (val[j] - mean) * rstd * g[i] + bta[i];
        orow[i] = o;
        brow[i] = (__bf16)o;
    }
}

// ---------------- host ----------------
extern "C" void kernel_launch(void* const* d_in, const int* in_sizes, int n_in,
                              void* d_out, int out_size, void* d_ws, size_t ws_size,
                              hipStream_t stream)
{
    const int*   ids  = (const int*)  d_in[0];
    const float* am   = (const float*)d_in[1];
    const float* tok  = (const float*)d_in[2];
    const float* pos  = (const float*)d_in[3];
    const float* Wq   = (const float*)d_in[4];
    const float* bq   = (const float*)d_in[5];
    const float* Wk   = (const float*)d_in[6];
    const float* bk   = (const float*)d_in[7];
    const float* Wv   = (const float*)d_in[8];
    const float* bv   = (const float*)d_in[9];
    const float* Wi   = (const float*)d_in[10];
    const float* bi   = (const float*)d_in[11];
    const float* Wo   = (const float*)d_in[12];
    const float* bo   = (const float*)d_in[13];
    const float* ln1g = (const float*)d_in[14];
    const float* ln1b = (const float*)d_in[15];
    const float* ln2g = (const float*)d_in[16];
    const float* ln2b = (const float*)d_in[17];
    const float* lnfg = (const float*)d_in[18];
    const float* lnfb = (const float*)d_in[19];
    const float* Wlm  = (const float*)d_in[20];
    float* out = (float*)d_out;

    // ---- workspace layout ----
    char* ws = (char*)d_ws;
    size_t off = 0;
    auto alloc = [&](size_t bytes) { void* p = ws + off; off += (bytes + 255) & ~(size_t)255; return p; };
    float*  h      = (float*) alloc((size_t)NTOK * HDIM * 4);
    float*  h1     = (float*) alloc((size_t)NTOK * HDIM * 4);
    float*  ctx    = (float*) alloc((size_t)NTOK * HDIM * 4);
    __bf16* h_bf   = (__bf16*)alloc((size_t)NTOK * HDIM * 2);
    __bf16* h1_bf  = (__bf16*)alloc((size_t)NTOK * HDIM * 2);
    __bf16* qb     = (__bf16*)alloc((size_t)NTOK * HDIM * 2);
    __bf16* kb     = (__bf16*)alloc((size_t)NTOK * HDIM * 2);
    __bf16* vb     = (__bf16*)alloc((size_t)NTOK * HDIM * 2);
    __bf16* mid_bf = (__bf16*)alloc((size_t)NTOK * IDIM * 2);
    __bf16* wq_b   = (__bf16*)alloc((size_t)NLAYER * HDIM * HDIM * 2);
    __bf16* wk_b   = (__bf16*)alloc((size_t)NLAYER * HDIM * HDIM * 2);
    __bf16* wv_b   = (__bf16*)alloc((size_t)NLAYER * HDIM * HDIM * 2);
    __bf16* wi_b   = (__bf16*)alloc((size_t)NLAYER * IDIM * HDIM * 2);
    __bf16* wo_b   = (__bf16*)alloc((size_t)NLAYER * HDIM * IDIM * 2);
    __bf16* wlm_b  = (__bf16*)alloc((size_t)VOCAB * HDIM * 2);

    dim3 blk256(256), blk128(128);

    // weights -> bf16 (deterministic, every call)
    auto cvt = [&](const float* src, __bf16* dst, size_t n) {
        cvt_kernel<<<dim3((unsigned)(n / (8 * 256))), blk256, 0, stream>>>(src, dst);
    };
    cvt(Wq,  wq_b,  (size_t)NLAYER * HDIM * HDIM);
    cvt(Wk,  wk_b,  (size_t)NLAYER * HDIM * HDIM);
    cvt(Wv,  wv_b,  (size_t)NLAYER * HDIM * HDIM);
    cvt(Wi,  wi_b,  (size_t)NLAYER * IDIM * HDIM);
    cvt(Wo,  wo_b,  (size_t)NLAYER * HDIM * IDIM);
    cvt(Wlm, wlm_b, (size_t)VOCAB * HDIM);

    dim3 gEmb(NTOK);
    dim3 gH(HDIM / 128, NTOK / 64);
    dim3 gI(IDIM / 128, NTOK / 64);
    dim3 gV(VOCAB / 128, NTOK / 64);
    dim3 gAttn(BATCH * NHEAD, SEQ / 64);
    dim3 gLN(NTOK);

    embed_kernel<<<gEmb, blk256, 0, stream>>>(ids, tok, pos, h, h_bf);

    for (int l = 0; l < NLAYER; ++l) {
        const __bf16* wq = wq_b + (size_t)l * HDIM * HDIM;
        const __bf16* wk = wk_b + (size_t)l * HDIM * HDIM;
        const __bf16* wv = wv_b + (size_t)l * HDIM * HDIM;
        const __bf16* wi = wi_b + (size_t)l * IDIM * HDIM;
        const __bf16* wo = wo_b + (size_t)l * HDIM * IDIM;

        gemm_kernel<0, __bf16><<<gH, blk256, 0, stream>>>(h_bf, wq, bq + l * HDIM, qb, NTOK, HDIM, HDIM);
        gemm_kernel<0, __bf16><<<gH, blk256, 0, stream>>>(h_bf, wk, bk + l * HDIM, kb, NTOK, HDIM, HDIM);
        gemm_kernel<0, __bf16><<<gH, blk256, 0, stream>>>(h_bf, wv, bv + l * HDIM, vb, NTOK, HDIM, HDIM);

        attn_kernel<<<gAttn, blk128, 0, stream>>>(qb, kb, vb, am, ctx);

        addln_kernel<<<gLN, blk256, 0, stream>>>(h, ctx, ln1g + l * HDIM, ln1b + l * HDIM, h1, h1_bf);

        gemm_kernel<1, __bf16><<<gI, blk256, 0, stream>>>(h1_bf, wi, bi + l * IDIM, mid_bf, NTOK, IDIM, HDIM);
        gemm_kernel<0, float ><<<gH, blk256, 0, stream>>>(mid_bf, wo, bo + l * HDIM, ctx, NTOK, HDIM, IDIM);

        addln_kernel<<<gLN, blk256, 0, stream>>>(h1, ctx, ln2g + l * HDIM, ln2b + l * HDIM, h, h_bf);
    }

    addln_kernel<<<gLN, blk256, 0, stream>>>(h, nullptr, lnfg, lnfb, h1, h1_bf);
    gemm_kernel<0, float><<<gV, blk256, 0, stream>>>(h1_bf, wlm_b, nullptr, out, NTOK, VOCAB, HDIM);
}
